// SSCLoss_62002147885205
// MI455X (gfx1250) — matrix operand order repaired and checked by reading, not compiled, not run on hardware
//
#include <hip/hip_runtime.h>
#include <hip/hip_bf16.h>

#define N_ROWS 4096
#define DDIM   1024
#define INV_T  2.0f     // 1/TEMPERATURE
#define EPS_N  1e-8f

typedef __attribute__((ext_vector_type(16))) _Float16 v16h;
typedef __attribute__((ext_vector_type(8)))  _Float16 v8h;
typedef __attribute__((ext_vector_type(8)))  float    v8f;

union V16H { v16h v; v8h h[2]; };

// ---------------------------------------------------------------------------
// Kernel 1: row L2-normalize (clamped) -> f16, copy labels to int32
// ---------------------------------------------------------------------------
__global__ void __launch_bounds__(128)
prep_kernel(const float* __restrict__ x, const int* __restrict__ tgt,
            _Float16* __restrict__ xn, int* __restrict__ lab) {
    __shared__ float red[128];
    const int row = blockIdx.x;
    const int t   = threadIdx.x;
    const float* xr = x + (size_t)row * DDIM;

    float vals[8];
    float ss = 0.f;
    #pragma unroll
    for (int q = 0; q < 8; ++q) {
        vals[q] = xr[t + q * 128];
        ss += vals[q] * vals[q];
    }
    red[t] = ss;
    __syncthreads();
    #pragma unroll
    for (int off = 64; off > 0; off >>= 1) {
        if (t < off) red[t] += red[t + off];
        __syncthreads();
    }
    const float inv = 1.0f / fmaxf(sqrtf(red[0]), EPS_N);
    _Float16* xo = xn + (size_t)row * DDIM;
    #pragma unroll
    for (int q = 0; q < 8; ++q) xo[t + q * 128] = (_Float16)(vals[q] * inv);
    if (t == 0) lab[row] = tgt[row];
}

// ---------------------------------------------------------------------------
// Two 16x16 f32 similarity tiles per call (shared A fragment, independent B
// streams -> two independent wmma dependence chains for latency hiding).
//   A rows = [r0, r0+16); B cols = [c0a, c0a+16) and [c0b, c0b+16)
// ---------------------------------------------------------------------------
__device__ __forceinline__ void sim_tile_pair(const _Float16* __restrict__ xn,
                                              int r0, int c0a, int c0b, int lane,
                                              v8f& ca, v8f& cb) {
    const int half = lane >> 4;     // 0: lanes 0-15, 1: lanes 16-31
    const int mr   = lane & 15;
    const _Float16* arow  = xn + (size_t)(r0  + mr) * DDIM;
    const _Float16* browa = xn + (size_t)(c0a + mr) * DDIM;
    const _Float16* browb = xn + (size_t)(c0b + mr) * DDIM;

    ca = (v8f){0.f, 0.f, 0.f, 0.f, 0.f, 0.f, 0.f, 0.f};
    cb = (v8f){0.f, 0.f, 0.f, 0.f, 0.f, 0.f, 0.f, 0.f};
    for (int k0 = 0; k0 < DDIM; k0 += 32) {
        // A fragment: K = k0+16h+[0..16), matching B
        V16H a;
        a.h[0] = *(const v8h*)(arow + k0 + 16 * half);
        a.h[1] = *(const v8h*)(arow + k0 + 16 * half + 8);
        // B fragments: N = mr, K = k0+16h+[0..16), contiguous 32B each
        const v16h ba = *(const v16h*)(browa + k0 + 16 * half);
        const v16h bb = *(const v16h*)(browb + k0 + 16 * half);
        ca = __builtin_amdgcn_wmma_f32_16x16x32_f16(
                 false, a.v, false, ba, (short)0, ca, false, false);
        cb = __builtin_amdgcn_wmma_f32_16x16x32_f16(
                 false, a.v, false, bb, (short)0, cb, false, false);
    }
}

// ---------------------------------------------------------------------------
// Pass A: neg[i] = sum_{label!=} e_ij ; t1[i] = sum_{label==, j!=i} 2*S_ij
// Block owns rows [16b,16b+16); 8 waves sweep the 256 column tiles, 2 at a time.
// ---------------------------------------------------------------------------
__global__ void __launch_bounds__(256)
passA_kernel(const _Float16* __restrict__ xn, const int* __restrict__ lab,
             float* __restrict__ neg, float* __restrict__ t1) {
    __shared__ float sNeg[8][16];
    __shared__ float sT1[8][16];
    const int r0   = blockIdx.x * 16;
    const int tid  = threadIdx.x;
    const int wave = tid >> 5;
    const int lane = tid & 31;
    const int half = lane >> 4;
    const int mr   = lane & 15;

    int labI[8];
    #pragma unroll
    for (int v = 0; v < 8; ++v) labI[v] = lab[r0 + v + 8 * half];

    float accNeg[8] = {0.f, 0.f, 0.f, 0.f, 0.f, 0.f, 0.f, 0.f};
    float accT1[8]  = {0.f, 0.f, 0.f, 0.f, 0.f, 0.f, 0.f, 0.f};

    for (int t = 0; t < (N_ROWS / 16) / 16; ++t) {       // 16 pair-iterations
        const int c0a = (wave + 16 * t) * 16;
        const int c0b = (wave + 8 + 16 * t) * 16;
        v8f cs[2];
        sim_tile_pair(xn, r0, c0a, c0b, lane, cs[0], cs[1]);
        const int c0s[2] = {c0a, c0b};
        #pragma unroll
        for (int p = 0; p < 2; ++p) {
            const int j    = c0s[p] + mr;
            const int labJ = lab[j];
            #pragma unroll
            for (int v = 0; v < 8; ++v) {
                const int   i    = r0 + v + 8 * half;    // C layout: row = v + 8*half
                const float sim2 = INV_T * cs[p][v];
                const float e    = __expf(sim2);
                const bool  same = (labI[v] == labJ);
                float nv = same ? 0.f : e;
                float tv = (same && (i != j)) ? sim2 : 0.f;
                #pragma unroll
                for (int m = 8; m >= 1; m >>= 1) {       // reduce over 16 cols
                    nv += __shfl_xor(nv, m, 32);
                    tv += __shfl_xor(tv, m, 32);
                }
                accNeg[v] += nv;
                accT1[v]  += tv;
            }
        }
    }

    if (mr == 0) {
        #pragma unroll
        for (int v = 0; v < 8; ++v) {
            sNeg[wave][v + 8 * half] = accNeg[v];
            sT1 [wave][v + 8 * half] = accT1[v];
        }
    }
    __syncthreads();
    if (tid < 16) {                                      // fixed-order combine
        float n = 0.f, t = 0.f;
        #pragma unroll
        for (int w = 0; w < 8; ++w) { n += sNeg[w][tid]; t += sT1[w][tid]; }
        neg[r0 + tid] = n;
        t1 [r0 + tid] = t;
    }
}

// ---------------------------------------------------------------------------
// Pass B: rl[i] = sum_{label==, j!=i} log(e_ij + neg_i)
// ---------------------------------------------------------------------------
__global__ void __launch_bounds__(256)
passB_kernel(const _Float16* __restrict__ xn, const int* __restrict__ lab,
             const float* __restrict__ neg, float* __restrict__ rl) {
    __shared__ float sL[8][16];
    const int r0   = blockIdx.x * 16;
    const int tid  = threadIdx.x;
    const int wave = tid >> 5;
    const int lane = tid & 31;
    const int half = lane >> 4;
    const int mr   = lane & 15;

    int   labI[8];
    float negI[8];
    #pragma unroll
    for (int v = 0; v < 8; ++v) {
        labI[v] = lab[r0 + v + 8 * half];
        negI[v] = neg[r0 + v + 8 * half];
    }

    float accL[8] = {0.f, 0.f, 0.f, 0.f, 0.f, 0.f, 0.f, 0.f};

    for (int t = 0; t < (N_ROWS / 16) / 16; ++t) {
        const int c0a = (wave + 16 * t) * 16;
        const int c0b = (wave + 8 + 16 * t) * 16;
        v8f cs[2];
        sim_tile_pair(xn, r0, c0a, c0b, lane, cs[0], cs[1]);
        const int c0s[2] = {c0a, c0b};
        #pragma unroll
        for (int p = 0; p < 2; ++p) {
            const int j    = c0s[p] + mr;
            const int labJ = lab[j];
            #pragma unroll
            for (int v = 0; v < 8; ++v) {
                const int   i    = r0 + v + 8 * half;
                const float sim2 = INV_T * cs[p][v];
                float lv = 0.f;
                if (labI[v] == labJ && i != j)
                    lv = __logf(__expf(sim2) + negI[v]);
                #pragma unroll
                for (int m = 8; m >= 1; m >>= 1) lv += __shfl_xor(lv, m, 32);
                accL[v] += lv;
            }
        }
    }

    if (mr == 0) {
        #pragma unroll
        for (int v = 0; v < 8; ++v) sL[wave][v + 8 * half] = accL[v];
    }
    __syncthreads();
    if (tid < 16) {
        float s = 0.f;
        #pragma unroll
        for (int w = 0; w < 8; ++w) s += sL[w][tid];
        rl[r0 + tid] = s;
    }
}

// ---------------------------------------------------------------------------
// Finalize: total = sum_i (rl[i] - t1[i]); cnt = n0(n0-1)+n1(n1-1); out = total/cnt
// ---------------------------------------------------------------------------
__global__ void __launch_bounds__(1024)
finalize_kernel(const float* __restrict__ rl, const float* __restrict__ t1,
                const int* __restrict__ lab, float* __restrict__ out) {
    __shared__ float sF[1024];
    __shared__ int   sC[1024];
    const int t = threadIdx.x;
    float s  = 0.f;
    int   c0 = 0;
    for (int i = t; i < N_ROWS; i += 1024) {
        s  += rl[i] - t1[i];
        c0 += (lab[i] == 0) ? 1 : 0;
    }
    sF[t] = s;
    sC[t] = c0;
    __syncthreads();
    #pragma unroll
    for (int off = 512; off > 0; off >>= 1) {
        if (t < off) { sF[t] += sF[t + off]; sC[t] += sC[t + off]; }
        __syncthreads();
    }
    if (t == 0) {
        const long long n0  = sC[0];
        const long long n1  = (long long)N_ROWS - n0;
        const long long cnt = n0 * (n0 - 1) + n1 * (n1 - 1);
        out[0] = sF[0] / (float)cnt;
    }
}

// ---------------------------------------------------------------------------
extern "C" void kernel_launch(void* const* d_in, const int* in_sizes, int n_in,
                              void* d_out, int out_size, void* d_ws, size_t ws_size,
                              hipStream_t stream) {
    const float* x   = (const float*)d_in[0];
    const int*   tgt = (const int*)d_in[1];

    char* ws = (char*)d_ws;
    _Float16* xn  = (_Float16*)ws;                                    // 8 MB
    int*      lab = (int*)(ws + (size_t)N_ROWS * DDIM * 2);           // 16 KB
    float*    neg = (float*)(ws + (size_t)N_ROWS * DDIM * 2 + (size_t)N_ROWS * 4);
    float*    t1  = neg + N_ROWS;
    float*    rl  = t1 + N_ROWS;
    float*    out = (float*)d_out;

    prep_kernel    <<<N_ROWS,      128,  0, stream>>>(x, tgt, xn, lab);
    passA_kernel   <<<N_ROWS / 16, 256,  0, stream>>>(xn, lab, neg, t1);
    passB_kernel   <<<N_ROWS / 16, 256,  0, stream>>>(xn, lab, neg, rl);
    finalize_kernel<<<1,           1024, 0, stream>>>(rl, t1, lab, out);
}
